// VectorQuantizer_26439818674354
// MI455X (gfx1250) — matrix operand order repaired and checked by reading, not compiled, not run on hardware
//
#include <hip/hip_runtime.h>

// VQ-VAE vector quantizer for MI455X (gfx1250, wave32).
// d = |e|^2 - 2 z.e via bf16x3-split WMMA (v_wmma_f32_16x16x32_bf16),
// codebook streamed through double-buffered LDS panels with async-to-LDS.
//
// Workspace layout (needs 32 MiB + 128 KiB):
//   [0,8M)   zbh   bf16 [16384][256]
//   [8,16M)  zbl
//   [16,24M) ebh
//   [24,32M) ebl
//   [32M, +64K)  enorm2 f32 [16384]
//   [+64K, +128K) idx    i32 [16384]

#define BETA 0.25f

typedef __attribute__((ext_vector_type(16))) __bf16 v16bf;
typedef __attribute__((ext_vector_type(8)))  __bf16 v8bf;
typedef __attribute__((ext_vector_type(8)))  float  v8f;
typedef __attribute__((ext_vector_type(4)))  int    v4i;

typedef __attribute__((address_space(1))) v4i gas_v4i;  // global (AS1)
typedef __attribute__((address_space(3))) v4i las_v4i;  // LDS (AS3)

#if __has_builtin(__builtin_amdgcn_global_load_async_to_lds_b128) && \
    __has_builtin(__builtin_amdgcn_s_wait_asynccnt)
#define HAS_ASYNC_LDS 1
#else
#define HAS_ASYNC_LDS 0
#endif

__device__ __forceinline__ void g2l_b128(__bf16* dst, const __bf16* src) {
#if HAS_ASYNC_LDS
  __builtin_amdgcn_global_load_async_to_lds_b128((gas_v4i*)src, (las_v4i*)dst,
                                                 0, 0);
#else
  *(v4i*)dst = *(const v4i*)src;
#endif
}

__device__ __forceinline__ void wait_async_lds() {
#if HAS_ASYNC_LDS
  __builtin_amdgcn_s_wait_asynccnt(0);
#endif
}

// ---------------------------------------------------------------------------
// Codebook prep: fp32 [16384][256] -> bf16 hi/lo + row |e|^2.
// One wave per row; lane handles 8 contiguous k (32B loads, coalesced).
__global__ __launch_bounds__(256) void vq_prep_codebook(
    const float* __restrict__ cb, __bf16* __restrict__ ebh,
    __bf16* __restrict__ ebl, float* __restrict__ enorm2) {
  const int wave = threadIdx.x >> 5;
  const int lane = threadIdx.x & 31;
  const int row  = blockIdx.x * 8 + wave;
  const float* src = cb + (size_t)row * 256 + lane * 8;
  float s = 0.0f;
  v8bf h, l;
#pragma unroll
  for (int j = 0; j < 8; ++j) {
    float x = src[j];
    __bf16 hb = (__bf16)x;
    h[j] = hb;
    l[j] = (__bf16)(x - (float)hb);
    s += x * x;
  }
  *(v8bf*)(ebh + (size_t)row * 256 + lane * 8) = h;
  *(v8bf*)(ebl + (size_t)row * 256 + lane * 8) = l;
#pragma unroll
  for (int off = 16; off >= 1; off >>= 1) s += __shfl_xor(s, off, 32);
  if (lane == 0) enorm2[row] = s;
}

// ---------------------------------------------------------------------------
// z prep: [16,256,32,32] NCHW -> bf16 hi/lo [M=16384][256], LDS-tiled transpose.
// grid = 16(b) * 32(h) * 8(dtile); block = 32x8.
__global__ __launch_bounds__(256) void vq_prep_z(
    const float* __restrict__ z, __bf16* __restrict__ zbh,
    __bf16* __restrict__ zbl) {
  __shared__ float tile[32][33];
  const int bt = blockIdx.x;
  const int dt = bt & 7, h = (bt >> 3) & 31, b = bt >> 8;
  const int tx = threadIdx.x & 31, ty = threadIdx.x >> 5;
#pragma unroll
  for (int j = 0; j < 4; ++j) {
    int dl = ty + j * 8;                       // local d
    int d = dt * 32 + dl;
    tile[dl][tx] = z[(((size_t)(b * 256 + d)) * 32 + h) * 32 + tx];
  }
  __syncthreads();
#pragma unroll
  for (int j = 0; j < 4; ++j) {
    int w = ty + j * 8;
    int m = b * 1024 + h * 32 + w;
    float x = tile[tx][w];                     // d_local = tx
    __bf16 hb = (__bf16)x;
    zbh[(size_t)m * 256 + dt * 32 + tx] = hb;
    zbl[(size_t)m * 256 + dt * 32 + tx] = (__bf16)(x - (float)hb);
  }
}

// ---------------------------------------------------------------------------
// GEMM + fused argmin. 256 blocks x 4 waves; wave owns 16 M rows.
// A fragments (8 K-chunks, hi+lo) resident in VGPRs, contiguous K order
// matching B: lane<16 -> K[0,16), lane>=16 -> K[16,32); VGPRs 0-3 hold
// K[kb,kb+8), VGPRs 4-7 hold K[kb+8,kb+16).
// B read from LDS panel in the same contiguous K order as A: lane<16 -> K[0,16),
// lane>=16 -> K[16,32), N = lane&15 (one 32B contiguous read).
__global__ __launch_bounds__(128) void vq_gemm_argmin(
    const __bf16* __restrict__ zbh, const __bf16* __restrict__ zbl,
    const __bf16* __restrict__ ebh, const __bf16* __restrict__ ebl,
    const float* __restrict__ enorm2, int* __restrict__ idx_ws,
    int* __restrict__ idx_out) {
  __shared__ __bf16 sBh[2][32 * 256];   // 16 KB each buffer
  __shared__ __bf16 sBl[2][32 * 256];   // total static LDS = 64 KB

  const int tid  = threadIdx.x;
  const int wave = tid >> 5;
  const int lane = tid & 31;
  const int l15  = lane & 15;
  const int m0   = blockIdx.x * 64 + wave * 16;

  // Resident A fragments (hi & lo) for all 8 K-chunks: 128 VGPRs.
  const int akb = (lane < 16) ? 0 : 16;
  v16bf Ah[8], Al[8];
  {
    const __bf16* zh = zbh + (size_t)(m0 + l15) * 256;
    const __bf16* zl = zbl + (size_t)(m0 + l15) * 256;
#pragma unroll
    for (int c = 0; c < 8; ++c) {
      union { v16bf v; v8bf h[2]; } th, tl;
      int k0 = c * 32 + akb;
      th.h[0] = *(const v8bf*)(zh + k0);
      th.h[1] = *(const v8bf*)(zh + k0 + 8);
      tl.h[0] = *(const v8bf*)(zl + k0);
      tl.h[1] = *(const v8bf*)(zl + k0 + 8);
      Ah[c] = th.v;
      Al[c] = tl.v;
    }
  }

  float dmin[8];
  int   imin[8];
#pragma unroll
  for (int i = 0; i < 8; ++i) { dmin[i] = __builtin_inff(); imin[i] = 0; }

  // Prime panel 0 (32 codebook rows -> LDS, 8x b128 per thread per array).
#pragma unroll
  for (int i = 0; i < 8; ++i) {
    int q = (i * 128 + tid) * 8;
    g2l_b128(&sBh[0][q], ebh + q);
    g2l_b128(&sBl[0][q], ebl + q);
  }
  wait_async_lds();
  __syncthreads();

  const int bkb = (lane < 16) ? 0 : 16;
  for (int p = 0; p < 512; ++p) {
    const int buf = p & 1;
    if (p + 1 < 512) {  // prefetch next panel into the other buffer
      const __bf16* sh = ebh + (size_t)(p + 1) * 32 * 256;
      const __bf16* sl = ebl + (size_t)(p + 1) * 32 * 256;
#pragma unroll
      for (int i = 0; i < 8; ++i) {
        int q = (i * 128 + tid) * 8;
        g2l_b128(&sBh[buf ^ 1][q], sh + q);
        g2l_b128(&sBl[buf ^ 1][q], sl + q);
      }
    }
#pragma unroll
    for (int s = 0; s < 2; ++s) {
      const __bf16* bh = &sBh[buf][(s * 16 + l15) * 256 + bkb];
      const __bf16* bl = &sBl[buf][(s * 16 + l15) * 256 + bkb];
      v8f acc = {0.f, 0.f, 0.f, 0.f, 0.f, 0.f, 0.f, 0.f};
#pragma unroll
      for (int c = 0; c < 8; ++c) {
        v16bf Bh = *(const v16bf*)(bh + c * 32);
        v16bf Bl = *(const v16bf*)(bl + c * 32);
        // bf16x3 split: hi*hi + hi*lo + lo*hi, fp32 accumulate.
        acc = __builtin_amdgcn_wmma_f32_16x16x32_bf16(false, Ah[c], false, Bh,
                                                      (short)0, acc, false, false);
        acc = __builtin_amdgcn_wmma_f32_16x16x32_bf16(false, Ah[c], false, Bl,
                                                      (short)0, acc, false, false);
        acc = __builtin_amdgcn_wmma_f32_16x16x32_bf16(false, Al[c], false, Bh,
                                                      (short)0, acc, false, false);
      }
      const int n = p * 32 + s * 16 + l15;
      const float en = enorm2[n];
#pragma unroll
      for (int i = 0; i < 8; ++i) {
        float d = __builtin_fmaf(-2.0f, acc[i], en);   // |e|^2 - 2 z.e
        if (d < dmin[i]) { dmin[i] = d; imin[i] = n; }
      }
    }
    wait_async_lds();
    __syncthreads();
  }

  // Argmin butterfly within each 16-lane half (ties -> smallest index).
#pragma unroll
  for (int off = 8; off >= 1; off >>= 1) {
#pragma unroll
    for (int i = 0; i < 8; ++i) {
      float od = __shfl_xor(dmin[i], off, 32);
      int   oi = __shfl_xor(imin[i], off, 32);
      if (od < dmin[i] || (od == dmin[i] && oi < imin[i])) {
        dmin[i] = od;
        imin[i] = oi;
      }
    }
  }
  // C/D layout: VGPR i holds row m0+i (lanes 0-15) and m0+8+i (lanes 16-31).
  if (l15 == 0) {
    const int mbase = m0 + ((lane >> 4) << 3);
#pragma unroll
    for (int i = 0; i < 8; ++i) {
      idx_ws[mbase + i]  = imin[i];
      idx_out[mbase + i] = imin[i];
    }
  }
}

// ---------------------------------------------------------------------------
// Gather z_q, write output NCHW, accumulate loss = 1.25 * mean((zq - z)^2).
__global__ __launch_bounds__(256) void vq_gather_loss(
    const float* __restrict__ z, const float* __restrict__ cb,
    const int* __restrict__ idx, float* __restrict__ out,
    float* __restrict__ loss) {
  const int o = blockIdx.x * 256 + threadIdx.x;
  const int w = o & 31, h = (o >> 5) & 31, c = (o >> 10) & 255, b = o >> 18;
  const int m = b * 1024 + h * 32 + w;
  const float q = cb[(size_t)idx[m] * 256 + c];
  const float diff = q - z[o];
  out[o] = q;
  float e = diff * diff;
#pragma unroll
  for (int off = 16; off >= 1; off >>= 1) e += __shfl_xor(e, off, 32);
  __shared__ float wsum[8];
  if ((threadIdx.x & 31) == 0) wsum[threadIdx.x >> 5] = e;
  __syncthreads();
  if (threadIdx.x == 0) {
    float s = 0.0f;
#pragma unroll
    for (int i = 0; i < 8; ++i) s += wsum[i];
    atomicAdd(loss, s * ((1.0f + BETA) / 4194304.0f));
  }
}

// ---------------------------------------------------------------------------
extern "C" void kernel_launch(void* const* d_in, const int* in_sizes, int n_in,
                              void* d_out, int out_size, void* d_ws,
                              size_t ws_size, hipStream_t stream) {
  (void)in_sizes; (void)n_in; (void)out_size; (void)ws_size;
  const float* z  = (const float*)d_in[0];   // [16,256,32,32] f32
  const float* cb = (const float*)d_in[1];   // [16384,256] f32

  char* ws = (char*)d_ws;
  __bf16* zbh = (__bf16*)(ws);
  __bf16* zbl = (__bf16*)(ws + (size_t)8 * 1024 * 1024);
  __bf16* ebh = (__bf16*)(ws + (size_t)16 * 1024 * 1024);
  __bf16* ebl = (__bf16*)(ws + (size_t)24 * 1024 * 1024);
  float* enorm2 = (float*)(ws + (size_t)32 * 1024 * 1024);
  int*   idxws  = (int*)(ws + (size_t)32 * 1024 * 1024 + 65536);

  float* out     = (float*)d_out;            // 4194304 f32
  float* loss    = out + 4194304;            // 1 f32
  int*   idx_out = (int*)(out + 4194305);    // 16384 i32

  vq_prep_codebook<<<2048, 256, 0, stream>>>(cb, ebh, ebl, enorm2);
  vq_prep_z<<<4096, 256, 0, stream>>>(z, zbh, zbl);
  vq_gemm_argmin<<<256, 128, 0, stream>>>(zbh, zbl, ebh, ebl, enorm2, idxws,
                                          idx_out);
  (void)hipMemsetAsync(loss, 0, sizeof(float), stream);
  vq_gather_loss<<<16384, 256, 0, stream>>>(z, cb, idxws, out, loss);
}
